// MultiHeadSeqAttention_55327768708022
// MI455X (gfx1250) — matrix operand order repaired and checked by reading, not compiled, not run on hardware
//
#include <hip/hip_runtime.h>

#define DEV __device__ __forceinline__

typedef __attribute__((ext_vector_type(16))) __bf16 v16bf;
typedef __attribute__((ext_vector_type(8)))  float  v8f;
typedef __attribute__((ext_vector_type(2)))  __bf16 v2bf;

union BFrag { v16bf v; unsigned int u[8]; };

#if __has_builtin(__builtin_amdgcn_global_load_async_to_lds_b128)
#define HAVE_ASYNC_LDS 1
#else
#define HAVE_ASYNC_LDS 0
#endif

#if HAVE_ASYNC_LDS
typedef int i32x4 __attribute__((vector_size(16)));
DEV void async_copy16(const void* g, void* l) {
  __builtin_amdgcn_global_load_async_to_lds_b128(
      (__attribute__((address_space(1))) i32x4*)g,
      (__attribute__((address_space(3))) i32x4*)l, 0, 0);
}
#if __has_builtin(__builtin_amdgcn_s_wait_asynccnt)
DEV void wait_async0() { __builtin_amdgcn_s_wait_asynccnt(0); }
DEV void wait_async2() { __builtin_amdgcn_s_wait_asynccnt(2); }
#else
DEV void wait_async0() { asm volatile("s_wait_asynccnt 0x0" ::: "memory"); }
DEV void wait_async2() { asm volatile("s_wait_asynccnt 0x2" ::: "memory"); }
#endif
#endif

// Native bf16 converts (lower to v_cvt_pk_bf16_f32 on gfx1250).
DEV unsigned short f2bf(float f) {
  union { __bf16 b; unsigned short u; } c;
  c.b = (__bf16)f;
  return c.u;
}
DEV unsigned int packcvt(float x, float y) {
  union { v2bf v; unsigned int u; } c;
  c.v.x = (__bf16)x;
  c.v.y = (__bf16)y;
  return c.u;
}
DEV float bf2f(unsigned short h) { return __uint_as_float(((unsigned int)h) << 16); }
DEV v8f wmma_bf16(v16bf a, v16bf b, v8f c) {
  return __builtin_amdgcn_wmma_f32_16x16x32_bf16(false, a, false, b, (short)0, c, false, false);
}
DEV float grp8_max(float v) {
  v = fmaxf(v, __shfl_xor(v, 1)); v = fmaxf(v, __shfl_xor(v, 2)); v = fmaxf(v, __shfl_xor(v, 4));
  return v;
}
DEV float grp8_sum(float v) {
  v += __shfl_xor(v, 1); v += __shfl_xor(v, 2); v += __shfl_xor(v, 4);
  return v;
}

// ---------------------------------------------------------------------------
// key_pe (64 x 1024 fp32, [d][s]) -> pet (1024 x 64 bf16, [s][d])
// ---------------------------------------------------------------------------
__global__ __launch_bounds__(256) void pe_transpose(const float* __restrict__ pe,
                                                    unsigned short* __restrict__ pet) {
  int i = blockIdx.x * 256 + threadIdx.x;  // 65536 elements
  int s = i >> 6, d = i & 63;
  pet[i] = f2bf(pe[(size_t)d * 1024 + s]);
}

// ---------------------------------------------------------------------------
// GEMM: Y = X @ W^T, X fp32 (B*L x 1024), W fp32 (1024 x 1024, row = out).
// VTRANS==false: head-split Y[((b*16+h)*L + m)*64 + d]
// VTRANS==true : head-split transposed Y[((b*16+h)*64 + d)*L + m]   (for V)
// L and VTRANS are compile-time: epilogue addressing folds to shifts.
// ---------------------------------------------------------------------------
template <int L, bool VTRANS>
__global__ __launch_bounds__(128) void gemm_qkv(const float* __restrict__ X,
                                                const float* __restrict__ W,
                                                unsigned short* __restrict__ Y) {
  __shared__ unsigned short As[64][40];
  __shared__ unsigned short Bs[64][40];
  const int bm0 = blockIdx.x * 64;
  const int n0  = blockIdx.y * 64;
  const int t = threadIdx.x, w = t >> 5, lane = t & 31;
  const int n = lane & 15, half = lane >> 4;
  const int srow = t >> 1, scol = (t & 1) * 16;

  const float* ax = X + (size_t)(bm0 + srow) * 1024 + scol;
  const float* bx = W + (size_t)(n0  + srow) * 1024 + scol;
  unsigned int* asd = (unsigned int*)&As[srow][scol];
  unsigned int* bsd = (unsigned int*)&Bs[srow][scol];

  v8f c0 = {}, c1 = {}, c2 = {}, c3 = {};

  for (int k0 = 0; k0 < 1024; k0 += 32) {
#pragma unroll
    for (int i = 0; i < 4; ++i) {
      float4 fa = *(const float4*)(ax + k0 + i * 4);
      asd[i * 2 + 0] = packcvt(fa.x, fa.y);
      asd[i * 2 + 1] = packcvt(fa.z, fa.w);
      float4 fb = *(const float4*)(bx + k0 + i * 4);
      bsd[i * 2 + 0] = packcvt(fb.x, fb.y);
      bsd[i * 2 + 1] = packcvt(fb.z, fb.w);
    }
    __syncthreads();

    BFrag a;
    const int arow = w * 16 + n;
#pragma unroll
    for (int j = 0; j < 8; ++j) {
      int kk = 16 * half + 2 * j;
      a.u[j] = *(const unsigned int*)&As[arow][kk];
    }
    BFrag b;
#pragma unroll
    for (int j = 0; j < 8; ++j) b.u[j] = *(const unsigned int*)&Bs[0 * 16 + n][16 * half + 2 * j];
    c0 = wmma_bf16(a.v, b.v, c0);
#pragma unroll
    for (int j = 0; j < 8; ++j) b.u[j] = *(const unsigned int*)&Bs[1 * 16 + n][16 * half + 2 * j];
    c1 = wmma_bf16(a.v, b.v, c1);
#pragma unroll
    for (int j = 0; j < 8; ++j) b.u[j] = *(const unsigned int*)&Bs[2 * 16 + n][16 * half + 2 * j];
    c2 = wmma_bf16(a.v, b.v, c2);
#pragma unroll
    for (int j = 0; j < 8; ++j) b.u[j] = *(const unsigned int*)&Bs[3 * 16 + n][16 * half + 2 * j];
    c3 = wmma_bf16(a.v, b.v, c3);
    __syncthreads();
  }

  v8f cc[4] = {c0, c1, c2, c3};
#pragma unroll
  for (int nb = 0; nb < 4; ++nb) {
#pragma unroll
    for (int r = 0; r < 8; ++r) {
      int gr = bm0 + w * 16 + r + 8 * half;
      int gc = n0 + nb * 16 + n;
      int bb = (gr >= L) ? 1 : 0;       // B == 2, avoid integer division
      int mr = gr - bb * L;
      int hh = gc >> 6, dd = gc & 63;
      if (VTRANS)
        Y[((size_t)(bb * 16 + hh) * 64 + dd) * (size_t)L + mr] = f2bf(cc[nb][r]);
      else
        Y[(((size_t)(bb * 16 + hh) * (size_t)L) + mr) * 64 + dd] = f2bf(cc[nb][r]);
    }
  }
}

#if HAVE_ASYNC_LDS
// Stage V^T tile: 64 d-rows x 32 cols (bf16) = 4 KB, 2 async b128 per thread.
DEV void stage_vt(const unsigned short* vt, unsigned short* buf, int m0, int kk, int t) {
  const int c0 = m0 + kk * 32;
#pragma unroll
  for (int rep = 0; rep < 2; ++rep) {
    int ch = t + rep * 128;              // 256 x 16B chunks
    int drow = ch >> 2, coff = (ch & 3) * 8;
    int cc = c0 + coff;
    if (cc > 3064) cc = 3064;            // weights there are exactly 0
    async_copy16(vt + (size_t)drow * 3072 + cc, buf + drow * 32 + coff);
  }
}
#endif

// ---------------------------------------------------------------------------
// Attention: one workgroup per (bh, 16-row query tile).
// ---------------------------------------------------------------------------
__global__ __launch_bounds__(128) void attn_kernel(const unsigned short* __restrict__ qb,
                                                   const unsigned short* __restrict__ kb,
                                                   const unsigned short* __restrict__ vbt,
                                                   const unsigned short* __restrict__ pet,
                                                   const float* __restrict__ span_val,
                                                   unsigned short* __restrict__ ctx) {
  __shared__ unsigned short sc[16][1056];      // 33 KB: scores -> attn weights
#if HAVE_ASYNC_LDS
  __shared__ unsigned short vts[3][64 * 32];   // 12 KB triple-buffered V^T tiles
#endif

  const int bh = blockIdx.x;          // 0..31
  const int m0 = blockIdx.y * 16;     // 0..2032
  const int b = bh >> 4, h = bh & 15;
  const unsigned short* q  = qb  + (size_t)bh * 2048 * 64;
  const unsigned short* k  = kb  + (size_t)bh * 3072 * 64;
  const unsigned short* vt = vbt + (size_t)bh * 64 * 3072;   // [d][row]

  const int t = threadIdx.x, w = t >> 5, lane = t & 31;
  const int n = lane & 15, half = lane >> 4;

  // Q fragments (rows m0..m0+15), reused across all column blocks.
  BFrag a0, a1;
  {
    const unsigned short* qrow = q + (size_t)(m0 + n) * 64;
#pragma unroll
    for (int j = 0; j < 8; ++j) {
      int kk = 16 * half + 2 * j;
      a0.u[j] = *(const unsigned int*)(qrow + kk);
      a1.u[j] = *(const unsigned int*)(qrow + 32 + kk);
    }
  }

  // ---- Phase 1: content scores over 65 absolute-l blocks ----
  for (int jb = w; jb < 65; jb += 4) {
    const unsigned short* krow = k + (size_t)(m0 + jb * 16 + n) * 64;
    if (jb + 4 < 65)
      __builtin_prefetch(k + (size_t)(m0 + (jb + 4) * 16 + n) * 64, 0, 0);
    BFrag b0, b1;
#pragma unroll
    for (int j = 0; j < 8; ++j) {
      int kk = 16 * half + 2 * j;
      b0.u[j] = *(const unsigned int*)(krow + kk);
      b1.u[j] = *(const unsigned int*)(krow + 32 + kk);
    }
    v8f c = {};
    c = wmma_bf16(a0.v, b0.v, c);
    c = wmma_bf16(a1.v, b1.v, c);
#pragma unroll
    for (int r = 0; r < 8; ++r) sc[r + 8 * half][jb * 16 + n] = f2bf(c[r]);
  }
  __syncthreads();

  // ---- Phase 2: positional scores (pet is [s][d] bf16), scatter-added ----
  for (int sb = w; sb < 64; sb += 4) {
    const unsigned short* prow = pet + (size_t)(sb * 16 + n) * 64;
    BFrag b0, b1;
#pragma unroll
    for (int j = 0; j < 8; ++j) {
      int kk = 16 * half + 2 * j;
      b0.u[j] = *(const unsigned int*)(prow + kk);
      b1.u[j] = *(const unsigned int*)(prow + 32 + kk);
    }
    v8f c = {};
    c = wmma_bf16(a0.v, b0.v, c);
    c = wmma_bf16(a1.v, b1.v, c);
#pragma unroll
    for (int r = 0; r < 8; ++r) {
      int m = r + 8 * half;
      int col = m + sb * 16 + n;
      sc[m][col] = f2bf(bf2f(sc[m][col]) + c[r]);
    }
  }
  __syncthreads();

  // ---- Phase 3: softmax + adaptive span mask + renorm (in place) ----
  {
    const int m = t >> 3, g = t & 7;   // 8 lanes per row
    float cur = span_val[h];
    cur = fminf(fmaxf(cur, 0.f), 1.f);
    const float curS = cur * 1024.0f;

    float mx = -3.0e38f;
    for (int s = g; s < 1024; s += 8) mx = fmaxf(mx, bf2f(sc[m][m + s]));
    mx = grp8_max(mx);

    float se = 0.f, sem = 0.f;
    for (int s = g; s < 1024; s += 8) {
      float e = __expf((bf2f(sc[m][m + s]) - mx) * 0.125f);
      float msk = fminf(fmaxf(((float)(s - 1023) + curS) * (1.0f / 32.0f) + 1.0f, 0.f), 1.f);
      se += e;
      sem += e * msk;
    }
    se = grp8_sum(se);
    sem = grp8_sum(sem);
    const float inv = 1.0f / (sem + 1e-8f * se);

    for (int l = g; l < 1056; l += 8) {
      int s = l - m;
      float av = 0.f;
      if (s >= 0 && s < 1024) {
        float e = __expf((bf2f(sc[m][l]) - mx) * 0.125f);
        float msk = fminf(fmaxf(((float)(s - 1023) + curS) * (1.0f / 32.0f) + 1.0f, 0.f), 1.f);
        av = e * msk * inv;
      }
      sc[m][l] = f2bf(av);
    }
  }
  __syncthreads();

  // ---- Phase 4: AV (wave w owns d-block w*16..w*16+15) ----
  const int d0 = w * 16;
  v8f c = {};
#if HAVE_ASYNC_LDS
  // Triple-buffered async pipeline, one barrier per step.
  stage_vt(vt, vts[0], m0, 0, t);
  for (int kk2 = 0; kk2 < 33; ++kk2) {
    if (kk2 + 1 < 33) stage_vt(vt, vts[(kk2 + 1) % 3], m0, kk2 + 1, t);
    if (kk2 + 1 < 33) wait_async2(); else wait_async0();   // tile kk2 landed (in-order)
    __syncthreads();                                       // visible to all waves

    const int l0 = kk2 * 32;
    BFrag a;
#pragma unroll
    for (int j = 0; j < 8; ++j) {
      int l = l0 + 16 * half + 2 * j;
      a.u[j] = *(const unsigned int*)&sc[n][l];
    }
    const unsigned short* vrow = &vts[kk2 % 3][(d0 + n) * 32];
    BFrag bv;
#pragma unroll
    for (int j = 0; j < 8; ++j) bv.u[j] = *(const unsigned int*)(vrow + 16 * half + 2 * j);
    c = wmma_bf16(a.v, bv.v, c);
  }
#else
  for (int kk2 = 0; kk2 < 33; ++kk2) {
    const int l0 = kk2 * 32;
    BFrag a;
#pragma unroll
    for (int j = 0; j < 8; ++j) {
      int l = l0 + 16 * half + 2 * j;
      a.u[j] = *(const unsigned int*)&sc[n][l];
    }
    const unsigned short* vrow = vt + (size_t)(d0 + n) * 3072;
    BFrag bv;
#pragma unroll
    for (int j = 0; j < 8; ++j) {
      int col = m0 + l0 + 16 * half + 2 * j;
      if (col > 3070) col = 3070;     // weights there are exactly 0
      bv.u[j] = *(const unsigned int*)(vrow + col);
    }
    c = wmma_bf16(a.v, bv.v, c);
  }
#endif
#pragma unroll
  for (int r = 0; r < 8; ++r) {
    int mm = r + 8 * half;
    ctx[((size_t)(b * 2048 + m0 + mm)) * 1024 + h * 64 + d0 + n] = f2bf(c[r]);
  }
}

// ---------------------------------------------------------------------------
// Final GEMM: out = ctx(bf16, 4096x1024) @ Wo^T(fp32) -> fp32.
// ---------------------------------------------------------------------------
__global__ __launch_bounds__(128) void gemm_out(const unsigned short* __restrict__ Cx,
                                                const float* __restrict__ Wo,
                                                float* __restrict__ out) {
  __shared__ unsigned short As[64][40];
  __shared__ unsigned short Bs[64][40];
  const int bm0 = blockIdx.x * 64;
  const int n0  = blockIdx.y * 64;
  const int t = threadIdx.x, w = t >> 5, lane = t & 31;
  const int n = lane & 15, half = lane >> 4;
  const int srow = t >> 1, scol = (t & 1) * 16;

  const unsigned short* ax = Cx + (size_t)(bm0 + srow) * 1024 + scol;
  const float* bx = Wo + (size_t)(n0 + srow) * 1024 + scol;
  unsigned int* asd = (unsigned int*)&As[srow][scol];
  unsigned int* bsd = (unsigned int*)&Bs[srow][scol];

  v8f c0 = {}, c1 = {}, c2 = {}, c3 = {};

  for (int k0 = 0; k0 < 1024; k0 += 32) {
#if HAVE_ASYNC_LDS
    async_copy16(ax + k0,     &As[srow][scol]);      // bf16 raw copy, 16B
    async_copy16(ax + k0 + 8, &As[srow][scol + 8]);
#else
#pragma unroll
    for (int i = 0; i < 4; ++i) {
      asd[i * 2 + 0] = *(const unsigned int*)(ax + k0 + i * 4);
      asd[i * 2 + 1] = *(const unsigned int*)(ax + k0 + i * 4 + 2);
    }
#endif
#pragma unroll
    for (int i = 0; i < 4; ++i) {
      float4 fb = *(const float4*)(bx + k0 + i * 4);
      bsd[i * 2 + 0] = packcvt(fb.x, fb.y);
      bsd[i * 2 + 1] = packcvt(fb.z, fb.w);
    }
#if HAVE_ASYNC_LDS
    wait_async0();
#endif
    __syncthreads();

    BFrag a;
    const int arow = w * 16 + n;
#pragma unroll
    for (int j = 0; j < 8; ++j) {
      int kk = 16 * half + 2 * j;
      a.u[j] = *(const unsigned int*)&As[arow][kk];
    }
    BFrag b;
#pragma unroll
    for (int j = 0; j < 8; ++j) b.u[j] = *(const unsigned int*)&Bs[0 * 16 + n][16 * half + 2 * j];
    c0 = wmma_bf16(a.v, b.v, c0);
#pragma unroll
    for (int j = 0; j < 8; ++j) b.u[j] = *(const unsigned int*)&Bs[1 * 16 + n][16 * half + 2 * j];
    c1 = wmma_bf16(a.v, b.v, c1);
#pragma unroll
    for (int j = 0; j < 8; ++j) b.u[j] = *(const unsigned int*)&Bs[2 * 16 + n][16 * half + 2 * j];
    c2 = wmma_bf16(a.v, b.v, c2);
#pragma unroll
    for (int j = 0; j < 8; ++j) b.u[j] = *(const unsigned int*)&Bs[3 * 16 + n][16 * half + 2 * j];
    c3 = wmma_bf16(a.v, b.v, c3);
    __syncthreads();
  }

  v8f cc[4] = {c0, c1, c2, c3};
#pragma unroll
  for (int nb = 0; nb < 4; ++nb) {
#pragma unroll
    for (int r = 0; r < 8; ++r) {
      int gr = bm0 + w * 16 + r + 8 * half;
      int gc = n0 + nb * 16 + n;
      out[(size_t)gr * 1024 + gc] = cc[nb][r];
    }
  }
}

// ---------------------------------------------------------------------------
extern "C" void kernel_launch(void* const* d_in, const int* in_sizes, int n_in,
                              void* d_out, int out_size, void* d_ws, size_t ws_size,
                              hipStream_t stream) {
  (void)in_sizes; (void)n_in; (void)out_size; (void)ws_size;
  const float* query   = (const float*)d_in[0];
  const float* key     = (const float*)d_in[1];
  const float* value   = (const float*)d_in[2];
  const float* key_pe  = (const float*)d_in[3];
  const float* Wq      = (const float*)d_in[4];
  const float* Wk      = (const float*)d_in[5];
  const float* Wv      = (const float*)d_in[6];
  const float* Wo      = (const float*)d_in[7];
  const float* span    = (const float*)d_in[8];

  unsigned short* qb  = (unsigned short*)d_ws;                   // 2*16*2048*64
  unsigned short* kb  = qb + (size_t)2 * 16 * 2048 * 64;         // 2*16*3072*64
  unsigned short* vbt = kb + (size_t)2 * 16 * 3072 * 64;         // 2*16*64*3072 (transposed)
  unsigned short* ctx = vbt + (size_t)2 * 16 * 3072 * 64 + 4096; // 2*2048*1024
  unsigned short* pet = ctx + (size_t)2 * 2048 * 1024;           // 1024*64

  pe_transpose<<<256, 256, 0, stream>>>(key_pe, pet);
  gemm_qkv<2048, false><<<dim3(4096 / 64, 16), 128, 0, stream>>>(query, Wq, qb);
  gemm_qkv<3072, false><<<dim3(6144 / 64, 16), 128, 0, stream>>>(key,   Wk, kb);
  gemm_qkv<3072, true ><<<dim3(6144 / 64, 16), 128, 0, stream>>>(value, Wv, vbt);
  attn_kernel<<<dim3(32, 128), 128, 0, stream>>>(qb, kb, vbt, pet, span, ctx);
  gemm_out<<<dim3(4096 / 64, 16), 128, 0, stream>>>(ctx, Wo, (float*)d_out);
}
